// LLM_618475291002
// MI455X (gfx1250) — compile-verified, matrix operand order checked by reading, not run on hardware
//
#include <hip/hip_runtime.h>
#include <math.h>

// ---------------------------------------------------------------------------
// GPT-2-ish forward (D=512,H=8,HD=64,L=6,T=2048,V=50257) for MI455X (gfx1250).
// Dense math: v_wmma_f32_16x16x32_f16 (f16 in, f32 accum).
// Tile movement: global_load_async_to_lds_b128 (ASYNCcnt) -> LDS -> WMMA frags,
// avoiding the VGPR round-trip / s_wait_loadcnt serialization.
// Attention: flash-style online softmax (scores never touch HBM).
// ---------------------------------------------------------------------------

typedef __attribute__((ext_vector_type(16))) _Float16 v16h;
typedef __attribute__((ext_vector_type(8)))  float    v8f;
typedef __attribute__((ext_vector_type(4)))  unsigned int u32x4;
typedef unsigned int u32;
typedef unsigned long long u64;

#define D_  512
#define H_  8
#define HD_ 64
#define L_  6
#define V_  50257
#define T_  2048
#define F_  2048   // 4*D

union FragA { v16h v; u32 u[8]; };
union FragC { v8f  v; float f[8]; };

// K-pair base for half-pair p (0..7) of a 16-bit WMMA A/B operand (ISA 7.12.2):
// VGPR0..3 hold K={0..7}, VGPR4..7 hold K={16..23} for lanes 0-15; +8 for 16-31.
// For a fixed lane the 8 dwords are contiguous at byte offs 0..15 and 32..47,
// so the compiler merges the gathers into two *_b128 loads.
__device__ __forceinline__ int kpair(int p, int lane) {
  return ((p < 4) ? (2 * p) : (2 * p + 8)) + ((lane & 16) ? 8 : 0);
}

// Reductions across a 16-lane half-wave (rows of a C-tile live on lanes 0-15
// or 16-31; xor masks 1,2,4,8 stay inside each half).
__device__ __forceinline__ float hg_max(float v) {
  #pragma unroll
  for (int m = 8; m >= 1; m >>= 1) v = fmaxf(v, __shfl_xor(v, m, 32));
  return v;
}
__device__ __forceinline__ float hg_sum(float v) {
  #pragma unroll
  for (int m = 8; m >= 1; m >>= 1) v += __shfl_xor(v, m, 32);
  return v;
}

// Async 16B/lane global->LDS copy (CDNA5 VGLOBAL op 98, ASYNCcnt-tracked).
// ldsOff is the wave-relative LDS byte offset (low 32 bits of a flat LDS ptr).
__device__ __forceinline__ void async_g2l_b128(u32 ldsOff, const void* g) {
  asm volatile("global_load_async_to_lds_b128 %0, %1, off"
               :: "v"(ldsOff), "v"((u64)(uintptr_t)g) : "memory");
}
__device__ __forceinline__ void wait_async0() {
  asm volatile("s_wait_asynccnt 0x0" ::: "memory");
}

// ---------------------------------------------------------------------------
__global__ void cvt_f32_to_f16(const float* __restrict__ src,
                               _Float16* __restrict__ dst, long n) {
  long i = (long)blockIdx.x * blockDim.x + threadIdx.x;
  long stride = (long)gridDim.x * blockDim.x;
  for (; i < n; i += stride) dst[i] = (_Float16)src[i];
}

__global__ void embed_kernel(const int* __restrict__ idx,
                             const float* __restrict__ tok,
                             const float* __restrict__ pos,
                             float* __restrict__ x) {
  int t = blockIdx.x;
  long row = idx[t];
  for (int j = threadIdx.x; j < D_; j += 256)
    x[(long)t * D_ + j] = tok[row * D_ + j] + pos[(long)t * D_ + j];
}

// LayerNorm over D=512, one row per block (256 threads, 2 elems each),
// writes f16 (feeds WMMA GEMMs directly).
__global__ __launch_bounds__(256) void ln_kernel(const float* __restrict__ x,
                                                 const float* __restrict__ s,
                                                 const float* __restrict__ b,
                                                 _Float16* __restrict__ out) {
  __shared__ float red[16];
  long t = blockIdx.x;
  const float* xr = x + t * D_;
  float v0 = xr[threadIdx.x], v1 = xr[threadIdx.x + 256];
  float sum = v0 + v1, sq = v0 * v0 + v1 * v1;
  #pragma unroll
  for (int m = 16; m >= 1; m >>= 1) {
    sum += __shfl_xor(sum, m, 32);
    sq  += __shfl_xor(sq,  m, 32);
  }
  int wv = threadIdx.x >> 5;
  if ((threadIdx.x & 31) == 0) { red[wv] = sum; red[8 + wv] = sq; }
  __syncthreads();
  if (threadIdx.x == 0) {
    float S = 0.f, Q = 0.f;
    for (int i = 0; i < 8; i++) { S += red[i]; Q += red[8 + i]; }
    red[0] = S; red[1] = Q;
  }
  __syncthreads();
  float mean = red[0] * (1.f / D_);
  float var  = red[1] * (1.f / D_) - mean * mean;
  float inv  = rsqrtf(var + 1e-5f);
  out[t * D_ + threadIdx.x] =
      (_Float16)((v0 - mean) * inv * s[threadIdx.x] + b[threadIdx.x]);
  out[t * D_ + threadIdx.x + 256] =
      (_Float16)((v1 - mean) * inv * s[threadIdx.x + 256] + b[threadIdx.x + 256]);
}

// ---------------------------------------------------------------------------
// WMMA GEMM: C[M,N](f32 accum) = A(f16 MxK rowmajor) * B(f16).
//  b_trans==0 : B is KxN row-major (N multiple of 64/128 on this path).
//  b_trans==1 : B stored N-major [N][K] (logits: B = tok_emb-f16). Rows n>=N
//               are loaded unguarded: garbage in B row n only reaches output
//               column n (independent dot products) and those columns are
//               discarded in the epilogue; reads stay inside the workspace.
// A tile and B^T tile move via global_load_async_to_lds_b128 (no VGPR stage).
// Block tile 128 x (32*NT): 8 waves in 4x2, each wave 2 x NT WMMA tiles, K=32.
#define BM 128
#define BK 32
#define SA 40   // padded LDS strides (halfs); 80B keeps 16B chunk alignment
#define SB 40

template <int NT>
__global__ __launch_bounds__(256) void gemm_wmma(
    const _Float16* __restrict__ A, const _Float16* __restrict__ B,
    int M, int N, int K, int b_trans,
    const float* __restrict__ bias, int gelu,
    const float* __restrict__ res,
    float* __restrict__ outF, _Float16* __restrict__ out16, int out16_trans) {
  constexpr int BN = 32 * NT;
  __shared__ __align__(16) _Float16 As[BM * SA];
  __shared__ __align__(16) _Float16 Bs[BN * SB];   // stored n-major [n][k]
  int n0 = blockIdx.x * BN;
  int m0 = blockIdx.y * BM;
  int tid  = threadIdx.x;
  int lane = tid & 31, wv = tid >> 5;
  int wm = wv & 3, wn = wv >> 2;     // 4x2 waves; each 32 x (NT*16) outputs

  u32 AsBase = (u32)(uintptr_t)(&As[0]);
  u32 BsBase = (u32)(uintptr_t)(&Bs[0]);

  FragC acc[2][NT];
  #pragma unroll
  for (int a = 0; a < 2; a++)
    #pragma unroll
    for (int bb = 0; bb < NT; bb++)
      #pragma unroll
      for (int j = 0; j < 8; j++) acc[a][bb].f[j] = 0.f;

  u32* Asu = (u32*)As;
  u32* Bsu = (u32*)Bs;

  for (int k0 = 0; k0 < K; k0 += BK) {
    // --- A tile: 128x32 halfs = 512 16B chunks, async direct to LDS ---
    #pragma unroll
    for (int i = 0; i < 2; i++) {
      int id = i * 256 + tid;
      int r = id >> 2, c8 = (id & 3) << 3;
      async_g2l_b128(AsBase + (u32)(r * SA + c8) * 2,
                     A + (long)(m0 + r) * K + k0 + c8);
    }
    // --- B tile ---
    if (b_trans) {  // N-major source: contiguous along k -> async b128
      #pragma unroll
      for (int i = 0; i < NT / 2; i++) {
        int id = i * 256 + tid;               // BN rows x 4 chunks
        int r = id >> 2, c8 = (id & 3) << 3;
        const _Float16* gp = B + (long)(n0 + r) * K + k0 + c8;
        __builtin_prefetch(gp + BK, 0, 0);    // speculative, next k-tile
        async_g2l_b128(BsBase + (u32)(r * SB + c8) * 2, gp);
      }
    } else {        // K-major source: b128 read + transposing b16 stores
      #pragma unroll
      for (int i = 0; i < NT / 2; i++) {
        int id = i * 256 + tid;               // 32 k-rows x (BN/8) n-chunks
        int r = id / (BN / 8), c8 = (id % (BN / 8)) << 3;
        u32x4 val = *(const u32x4*)(B + (long)(k0 + r) * N + n0 + c8);
        const _Float16* hp = (const _Float16*)&val;
        #pragma unroll
        for (int j = 0; j < 8; j++) Bs[(c8 + j) * SB + r] = hp[j];
      }
    }
    wait_async0();
    __syncthreads();

    FragA a[2], b[NT];
    #pragma unroll
    for (int mt = 0; mt < 2; mt++) {
      int mrow = wm * 32 + mt * 16 + (lane & 15);
      #pragma unroll
      for (int p = 0; p < 8; p++)
        a[mt].u[p] = Asu[(mrow * SA + kpair(p, lane)) >> 1];
    }
    #pragma unroll
    for (int nt = 0; nt < NT; nt++) {
      int nrow = wn * (NT * 16) + nt * 16 + (lane & 15);
      #pragma unroll
      for (int p = 0; p < 8; p++)
        b[nt].u[p] = Bsu[(nrow * SB + kpair(p, lane)) >> 1];
    }
    #pragma unroll
    for (int mt = 0; mt < 2; mt++)
      #pragma unroll
      for (int nt = 0; nt < NT; nt++)
        acc[mt][nt].v = __builtin_amdgcn_wmma_f32_16x16x32_f16(
            false, a[mt].v, false, b[nt].v, (short)0, acc[mt][nt].v, false, false);
    __syncthreads();   // WMMAs force dscnt drain first; async writes of the
                       // next iteration start only after this barrier
  }

  // --- epilogue: +bias, erf-GELU, +residual, f32/f16 (optionally transposed)
  int rbase = m0 + wm * 32;
  int cbase = n0 + wn * (NT * 16);
  int ro = (lane >> 4) << 3;               // C-tile row offset: 0 or 8
  #pragma unroll
  for (int mt = 0; mt < 2; mt++) {
    #pragma unroll
    for (int nt = 0; nt < NT; nt++) {
      int c = cbase + nt * 16 + (lane & 15);
      if (c >= N) continue;
      float bi = bias ? bias[c] : 0.f;
      #pragma unroll
      for (int j = 0; j < 8; j++) {
        int r = rbase + mt * 16 + ro + j;
        float v = acc[mt][nt].f[j] + bi;
        if (gelu) v = 0.5f * v * (1.f + erff(v * 0.70710678118654752f));
        if (res)  v += res[(long)r * N + c];
        if (outF) outF[(long)r * N + c] = v;
        if (out16) {
          if (out16_trans) out16[(long)c * M + r] = (_Float16)v;
          else             out16[(long)r * N + c] = (_Float16)v;
        }
      }
    }
  }
}

// ---------------------------------------------------------------------------
// Flash attention (causal). Block = 4 waves; each wave owns a 16-row query
// strip of one head; 32-key tiles; Q.K^T and P.V on WMMA; online softmax with
// half-wave shuffle reductions; P goes through per-wave LDS to convert the
// C/D layout into an A-operand layout. Loop bounds are wave-uniform so EXEC
// is all-ones at every WMMA (ISA 7.12 requirement).
__global__ __launch_bounds__(128) void attn_kernel(
    const _Float16* __restrict__ q16, const _Float16* __restrict__ k16,
    const _Float16* __restrict__ vT16, _Float16* __restrict__ o16) {
  __shared__ __align__(16) _Float16 Pls[4][16 * 40];
  int lane = threadIdx.x & 31, wv = threadIdx.x >> 5;
  int h   = blockIdx.y;
  int q0  = blockIdx.x * 64 + wv * 16;
  int hd0 = h * HD_;

  // Q fragments (16x64 -> two K-chunks of 32), loaded once.
  FragA aQ[2];
  {
    const _Float16* qrow = q16 + (long)(q0 + (lane & 15)) * D_ + hd0;
    #pragma unroll
    for (int kc = 0; kc < 2; kc++)
      #pragma unroll
      for (int p = 0; p < 8; p++)
        aQ[kc].u[p] = *(const u32*)(qrow + kc * 32 + kpair(p, lane));
  }

  FragC Oacc[4];
  #pragma unroll
  for (int t = 0; t < 4; t++)
    #pragma unroll
    for (int j = 0; j < 8; j++) Oacc[t].f[j] = 0.f;
  float mrow[8], lrow[8];
  #pragma unroll
  for (int j = 0; j < 8; j++) { mrow[j] = -3e38f; lrow[j] = 0.f; }

  _Float16* P = &Pls[wv][0];
  u32* Pu = (u32*)P;

  for (int k0 = 0; k0 <= q0 + 15; k0 += 32) {
    // ---- scores S = (Q K^T) * 1/sqrt(64), two 16x16 n-halves ----
    FragC S[2];
    #pragma unroll
    for (int nh = 0; nh < 2; nh++) {
      #pragma unroll
      for (int j = 0; j < 8; j++) S[nh].f[j] = 0.f;
      const _Float16* krow = k16 + (long)(k0 + nh * 16 + (lane & 15)) * D_ + hd0;
      #pragma unroll
      for (int kc = 0; kc < 2; kc++) {
        FragA bK;
        #pragma unroll
        for (int p = 0; p < 8; p++)
          bK.u[p] = *(const u32*)(krow + kc * 32 + kpair(p, lane));
        S[nh].v = __builtin_amdgcn_wmma_f32_16x16x32_f16(
            false, aQ[kc].v, false, bK.v, (short)0, S[nh].v, false, false);
      }
      int key = k0 + nh * 16 + (lane & 15);
      #pragma unroll
      for (int j = 0; j < 8; j++) {
        int qr = q0 + j + ((lane >> 4) << 3);
        float sc = S[nh].f[j] * 0.125f;                 // 1/sqrt(HD)
        S[nh].f[j] = (key <= qr) ? sc : -3e38f;         // causal mask
      }
    }
    // ---- online softmax update ----
    #pragma unroll
    for (int j = 0; j < 8; j++) {
      float rm    = hg_max(fmaxf(S[0].f[j], S[1].f[j]));
      float mnew  = fmaxf(mrow[j], rm);
      float alpha = __expf(mrow[j] - mnew);
      float p0 = __expf(S[0].f[j] - mnew);
      float p1 = __expf(S[1].f[j] - mnew);
      lrow[j] = lrow[j] * alpha + hg_sum(p0 + p1);
      mrow[j] = mnew;
      #pragma unroll
      for (int t = 0; t < 4; t++) Oacc[t].f[j] *= alpha;
      int prow = j + ((lane >> 4) << 3);
      P[prow * 40 + (lane & 15)]      = (_Float16)p0;
      P[prow * 40 + 16 + (lane & 15)] = (_Float16)p1;
    }
    // per-wave LDS RAW: DS pipe is in-order per wave; explicit drain anyway
    asm volatile("s_wait_dscnt 0x0" ::: "memory");
    // ---- P (16x32) as A operand; accumulate O += P * V ----
    FragA aP;
    #pragma unroll
    for (int p = 0; p < 8; p++)
      aP.u[p] = Pu[((lane & 15) * 40 + kpair(p, lane)) >> 1];
    #pragma unroll
    for (int t = 0; t < 4; t++) {
      FragA bV;
      const _Float16* vrow = vT16 + (long)(hd0 + t * 16 + (lane & 15)) * T_ + k0;
      #pragma unroll
      for (int p = 0; p < 8; p++)
        bV.u[p] = *(const u32*)(vrow + kpair(p, lane));
      Oacc[t].v = __builtin_amdgcn_wmma_f32_16x16x32_f16(
          false, aP.v, false, bV.v, (short)0, Oacc[t].v, false, false);
    }
  }
  // ---- normalize and store f16 ----
  #pragma unroll
  for (int t = 0; t < 4; t++)
    #pragma unroll
    for (int j = 0; j < 8; j++) {
      int r = q0 + j + ((lane >> 4) << 3);
      o16[(long)r * D_ + hd0 + t * 16 + (lane & 15)] =
          (_Float16)(Oacc[t].f[j] / lrow[j]);
    }
}

// ---------------------------------------------------------------------------
extern "C" void kernel_launch(void* const* d_in, const int* in_sizes, int n_in,
                              void* d_out, int out_size, void* d_ws, size_t ws_size,
                              hipStream_t stream) {
  const int*   idx     = (const int*)  d_in[0];
  const float* tok_emb = (const float*)d_in[1];
  const float* pos_emb = (const float*)d_in[2];
  const float* ln1_s   = (const float*)d_in[3];
  const float* ln1_b   = (const float*)d_in[4];
  const float* Wq      = (const float*)d_in[5];
  const float* Wk      = (const float*)d_in[6];
  const float* Wv      = (const float*)d_in[7];
  const float* Wo      = (const float*)d_in[8];
  const float* ln2_s   = (const float*)d_in[9];
  const float* ln2_b   = (const float*)d_in[10];
  const float* W1      = (const float*)d_in[11];
  const float* b1      = (const float*)d_in[12];
  const float* W2      = (const float*)d_in[13];
  const float* b2      = (const float*)d_in[14];
  const float* lnf_s   = (const float*)d_in[15];
  const float* lnf_b   = (const float*)d_in[16];
  float* out = (float*)d_out;

  char* ws = (char*)d_ws;
  size_t off = 0;
  auto carve = [&](size_t bytes) -> char* {
    char* p = ws + off;
    off += (bytes + 255) & ~(size_t)255;
    return p;
  };
  _Float16* tok16 = (_Float16*)carve((size_t)V_ * D_ * 2);
  _Float16* Wq16  = (_Float16*)carve((size_t)L_ * D_ * D_ * 2);
  _Float16* Wk16  = (_Float16*)carve((size_t)L_ * D_ * D_ * 2);
  _Float16* Wv16  = (_Float16*)carve((size_t)L_ * D_ * D_ * 2);
  _Float16* Wo16  = (_Float16*)carve((size_t)L_ * D_ * D_ * 2);
  _Float16* W116  = (_Float16*)carve((size_t)L_ * D_ * F_ * 2);
  _Float16* W216  = (_Float16*)carve((size_t)L_ * F_ * D_ * 2);
  float*    x     = (float*)   carve((size_t)T_ * D_ * 4);
  _Float16* h16   = (_Float16*)carve((size_t)T_ * D_ * 2);
  _Float16* q16   = (_Float16*)carve((size_t)T_ * D_ * 2);
  _Float16* k16   = (_Float16*)carve((size_t)T_ * D_ * 2);
  _Float16* vT16  = (_Float16*)carve((size_t)D_ * T_ * 2);  // [D][T]
  _Float16* o16   = (_Float16*)carve((size_t)T_ * D_ * 2);
  _Float16* g16   = (_Float16*)carve((size_t)T_ * F_ * 2);

  auto cvt = [&](const float* s, _Float16* d, long n) {
    cvt_f32_to_f16<<<2048, 256, 0, stream>>>(s, d, n);
  };
  cvt(tok_emb, tok16, (long)V_ * D_);
  cvt(Wq, Wq16, (long)L_ * D_ * D_);
  cvt(Wk, Wk16, (long)L_ * D_ * D_);
  cvt(Wv, Wv16, (long)L_ * D_ * D_);
  cvt(Wo, Wo16, (long)L_ * D_ * D_);
  cvt(W1, W116, (long)L_ * D_ * F_);
  cvt(W2, W216, (long)L_ * F_ * D_);

  embed_kernel<<<T_, 256, 0, stream>>>(idx, tok_emb, pos_emb, x);

  // NT=4 (BN=128) for wide GEMMs (higher WMMA/byte), NT=2 (BN=64) otherwise
  // to keep grid occupancy for the small projections.
  auto gemm = [&](const _Float16* A, const _Float16* B, int M, int N, int K,
                  int b_trans, const float* bias, int gelu, const float* res,
                  float* outF, _Float16* o16p, int o16t) {
    if (N >= 1024) {
      dim3 g((N + 127) / 128, M / BM);
      gemm_wmma<4><<<g, 256, 0, stream>>>(A, B, M, N, K, b_trans, bias, gelu,
                                          res, outF, o16p, o16t);
    } else {
      dim3 g((N + 63) / 64, M / BM);
      gemm_wmma<2><<<g, 256, 0, stream>>>(A, B, M, N, K, b_trans, bias, gelu,
                                          res, outF, o16p, o16t);
    }
  };

  for (int l = 0; l < L_; l++) {
    const _Float16* wq  = Wq16 + (size_t)l * D_ * D_;
    const _Float16* wk  = Wk16 + (size_t)l * D_ * D_;
    const _Float16* wvv = Wv16 + (size_t)l * D_ * D_;
    const _Float16* wo  = Wo16 + (size_t)l * D_ * D_;
    const _Float16* w1  = W116 + (size_t)l * D_ * F_;
    const _Float16* w2  = W216 + (size_t)l * F_ * D_;

    ln_kernel<<<T_, 256, 0, stream>>>(x, ln1_s + l * D_, ln1_b + l * D_, h16);
    gemm(h16, wq, T_, D_, D_, 0, nullptr, 0, nullptr, nullptr, q16, 0);
    gemm(h16, wk, T_, D_, D_, 0, nullptr, 0, nullptr, nullptr, k16, 0);
    gemm(h16, wvv, T_, D_, D_, 0, nullptr, 0, nullptr, nullptr, vT16, 1); // V^T
    attn_kernel<<<dim3(T_ / 64, H_), 128, 0, stream>>>(q16, k16, vT16, o16);
    gemm(o16, wo, T_, D_, D_, 0, nullptr, 0, x, x, nullptr, 0);           // +res
    ln_kernel<<<T_, 256, 0, stream>>>(x, ln2_s + l * D_, ln2_b + l * D_, h16);
    gemm(h16, w1, T_, F_, D_, 0, b1 + (size_t)l * F_, 1, nullptr, nullptr, g16, 0);
    gemm(g16, w2, T_, D_, F_, 0, b2 + (size_t)l * D_, 0, x, x, nullptr, 0);
  }

  ln_kernel<<<T_, 256, 0, stream>>>(x, lnf_s, lnf_b, h16);
  // logits = xln @ tok_emb^T  (B^T async path; N=50257 guarded in epilogue)
  gemm(h16, tok16, T_, V_, D_, 1, nullptr, 0, nullptr, out, nullptr, 0);
}
